// Attention_3917010174053
// MI455X (gfx1250) — matrix operand order repaired and checked by reading, not compiled, not run on hardware
//
#include <hip/hip_runtime.h>

// ---------------------------------------------------------------------------
// MI455X / gfx1250 fused attention: bf16 WMMA (16x16x32), fp32 accum,
// async-DMA (GLOBAL_LOAD_ASYNC_TO_LDS_B128) staging for GEMM tiles and the
// attention K-block; flash-style online softmax with 128-key steps.
// ---------------------------------------------------------------------------

typedef __attribute__((ext_vector_type(8)))  __bf16 v8bf;
typedef __attribute__((ext_vector_type(16))) __bf16 v16bf;
typedef __attribute__((ext_vector_type(8)))  float  v8f;

#define WMMA_BF16(a, b, c)                                                     \
  __builtin_amdgcn_wmma_f32_16x16x32_bf16(false, (a), false, (b), (short)0,    \
                                          (c), false, false)

static __device__ __forceinline__ __bf16 f2bf(float f) {
  // round-to-nearest-even fp32 -> bf16
  unsigned u = __builtin_bit_cast(unsigned, f);
  unsigned r = (u + 0x7FFFu + ((u >> 16) & 1u)) >> 16;
  unsigned short s = (unsigned short)r;
  return __builtin_bit_cast(__bf16, s);
}

static __device__ __forceinline__ v16bf combine16(v8bf lo, v8bf hi) {
  return __builtin_shufflevector(lo, hi, 0, 1, 2, 3, 4, 5, 6, 7, 8, 9, 10, 11,
                                 12, 13, 14, 15);
}

// Async DMA: 16 bytes global -> LDS, tracked by ASYNCcnt (no VGPR data path).
// VDST = per-lane LDS byte address (flat-address low 32 bits == LDS offset),
// VADDR = per-lane 64-bit global address, GV mode (no SADDR).
static __device__ __forceinline__ void async_cp16(void* lds, const void* g) {
  unsigned l = (unsigned)(unsigned long long)lds;
  unsigned long long ga = (unsigned long long)g;
  asm volatile("global_load_async_to_lds_b128 %0, %1, off"
               :: "v"(l), "v"(ga) : "memory");
}
static __device__ __forceinline__ void wait_async0() {
  asm volatile("s_wait_asynccnt 0x0" ::: "memory");
}

// ---------------------------------------------------------------------------
// fp32 -> bf16 elementwise convert (grid-stride)
// ---------------------------------------------------------------------------
__global__ void f32_to_bf16_kernel(const float* __restrict__ in,
                                   __bf16* __restrict__ out, long long n) {
  long long i = (long long)blockIdx.x * blockDim.x + threadIdx.x;
  long long stride = (long long)gridDim.x * blockDim.x;
  for (; i < n; i += stride) out[i] = f2bf(in[i]);
}

// ---------------------------------------------------------------------------
// GEMM: C[M,N] = A[M,K] (bf16 rm) * W[N,K]^T (bf16 rm) + bias
// Block: 256 threads (8 waves), workgroup tile 128(M) x 128(N), K-step 32.
// A/B tiles double-buffered in LDS via async DMA; waves consume via
// ds_load_b128 fragment reads. Wave tile 32(M) x 64(N) = 8 WMMA per step.
// LDS row stride 48 elems (96 B): 16B-aligned chunks, staggered banks.
// ---------------------------------------------------------------------------
template <bool STORE_BF16>
__global__ __launch_bounds__(256) void gemm_bias_wmma(
    const __bf16* __restrict__ A, const __bf16* __restrict__ W,
    const float* __restrict__ bias, void* __restrict__ Cout, int M, int N,
    int K) {
  __shared__ __align__(16) __bf16 Abuf[2][128][48];
  __shared__ __align__(16) __bf16 Bbuf[2][128][48];

  const int tid = threadIdx.x;
  const int lane = tid & 31;
  const int wave = tid >> 5;  // 0..7
  const int nl = lane & 15;
  const int hf = lane >> 4;
  const int mw = (wave & 3) * 32;
  const int nw = (wave >> 2) * 64;
  const int mBase = blockIdx.y * 128;
  const int nBase = blockIdx.x * 128;

  // Staging: thread t moves 32 B of the A tile and 32 B of the B tile.
  const int sr = tid >> 1;        // tile row 0..127
  const int sh = (tid & 1) * 16;  // element offset 0 / 16 within 32-wide k
  const __bf16* gA = A + (size_t)(mBase + sr) * K + sh;
  const __bf16* gW = W + (size_t)(nBase + sr) * K + sh;

#define STAGE(buf, k0)                                                         \
  do {                                                                         \
    async_cp16(&Abuf[buf][sr][sh], gA + (k0));                                 \
    async_cp16(&Abuf[buf][sr][sh + 8], gA + (k0) + 8);                         \
    async_cp16(&Bbuf[buf][sr][sh], gW + (k0));                                 \
    async_cp16(&Bbuf[buf][sr][sh + 8], gW + (k0) + 8);                         \
  } while (0)

  v8f acc[2][4] = {};

  STAGE(0, 0);
  wait_async0();
  __syncthreads();

  int cur = 0;
  for (int k0 = 0; k0 < K; k0 += 32) {
    if (k0 + 32 < K) STAGE(cur ^ 1, k0 + 32);  // overlap DMA with WMMAs

    const __bf16* Ab = &Abuf[cur][0][0];
    const __bf16* Bb = &Bbuf[cur][0][0];
    const __bf16* ap0 = Ab + (mw + nl) * 48 + hf * 8;
    const __bf16* ap1 = Ab + (mw + 16 + nl) * 48 + hf * 8;
    v16bf a0 = combine16(*(const v8bf*)ap0, *(const v8bf*)(ap0 + 16));
    v16bf a1 = combine16(*(const v8bf*)ap1, *(const v8bf*)(ap1 + 16));
#pragma unroll
    for (int j = 0; j < 4; j++) {
      const __bf16* bp = Bb + (nw + 16 * j + nl) * 48 + hf * 8;
      v16bf bf = combine16(*(const v8bf*)bp, *(const v8bf*)(bp + 16));
      acc[0][j] = WMMA_BF16(a0, bf, acc[0][j]);
      acc[1][j] = WMMA_BF16(a1, bf, acc[1][j]);
    }

    wait_async0();     // own DMA into next buffer landed
    __syncthreads();   // everyone done reading cur + all DMA visible
    cur ^= 1;
  }
#undef STAGE

  // C/D layout: lane L holds (row = v + 8*(L/16), col = L%16) in vgpr v.
#pragma unroll
  for (int j = 0; j < 4; j++) {
    int col = nBase + nw + 16 * j + nl;
    float bv = bias[col];
#pragma unroll
    for (int i = 0; i < 2; i++) {
#pragma unroll
      for (int v = 0; v < 8; v++) {
        int row = mBase + mw + 16 * i + v + 8 * hf;
        float val = acc[i][j][v] + bv;
        if (STORE_BF16)
          ((__bf16*)Cout)[(size_t)row * N + col] = f2bf(val);
        else
          ((float*)Cout)[(size_t)row * N + col] = val;
      }
    }
  }
}

// ---------------------------------------------------------------------------
// Fused flash attention (non-causal: reference applies full softmax over T).
// qkv: [B*T, 3072] bf16, chunk order (k | q | v), head h at offset h*64.
// Grid: (T/64, NH, B); block 128 (4 waves). Wave handles 16 Q rows.
// 128-key steps: K block async-DMA'd to LDS (shared by all 4 waves),
// V block transposed into LDS, S in registers, P exchanged via LDS, online softmax with
// amortized half-wave shuffle reductions.
// Output written flat as [B, nh, T, hs] -> directly reused as [B,T,C]
// (faithful to the reference's discarded-transpose reshape).
// ---------------------------------------------------------------------------
__global__ __launch_bounds__(128) void attn_fused_wmma(
    const __bf16* __restrict__ qkv, __bf16* __restrict__ out) {
  const int T = 2048, C3 = 3072, NH = 16, HS = 64;
  __shared__ __align__(16) __bf16 Kbuf[128][72];   // K block, row-major
  __shared__ __align__(16) __bf16 Vt[64][136];     // V block, transposed
  __shared__ __align__(16) __bf16 Pl[4][16][136];  // per-wave P tile (16x128)

  const int tid = threadIdx.x;
  const int lane = tid & 31;
  const int wave = tid >> 5;
  const int nl = lane & 15;
  const int hf = lane >> 4;
  const int b = blockIdx.z, h = blockIdx.y;
  const int qbase = blockIdx.x * 64 + wave * 16;

  const __bf16* base = qkv + (size_t)b * T * C3;

  // Q fragments for this wave's 16 rows (K-dim = hs = 64 -> two 16x32 frags)
  const __bf16* qp = base + (size_t)(qbase + nl) * C3 + 1024 + h * HS;
  v16bf aq0 = combine16(*(const v8bf*)(qp + hf * 8),
                        *(const v8bf*)(qp + hf * 8 + 16));
  v16bf aq1 = combine16(*(const v8bf*)(qp + 32 + hf * 8),
                        *(const v8bf*)(qp + 32 + hf * 8 + 16));

  float mrun[8], lrun[8];
  v8f o[4] = {};
#pragma unroll
  for (int v = 0; v < 8; v++) {
    mrun[v] = -1e30f;
    lrun[v] = 0.f;
  }
  const float scale = 0.125f;  // hs^-0.5

  for (int tk = 0; tk < T; tk += 128) {
    // Prefetch next key block's K and V rows (global_prefetch_b8)
    if (tk + 128 < T) {
      const __bf16* nxt = base + (size_t)(tk + 128 + tid) * C3 + h * HS;
      __builtin_prefetch(nxt, 0, 0);
      __builtin_prefetch(nxt + 2048, 0, 0);
    }

    __syncthreads();  // previous iteration's Kbuf/Vt readers done

    // Async-DMA the K block into LDS (one row per thread, 8 x 16 B).
    {
      const __bf16* krow = base + (size_t)(tk + tid) * C3 + h * HS;
#pragma unroll
      for (int c = 0; c < 64; c += 8) async_cp16(&Kbuf[tid][c], krow + c);
    }
    // Stage V^T (one V row per thread; transpose via register scatter),
    // overlapped with the K DMA above.
    {
      const __bf16* vrow = base + (size_t)(tk + tid) * C3 + 2048 + h * HS;
#pragma unroll
      for (int c = 0; c < 64; c += 8) {
        v8bf x = *(const v8bf*)(vrow + c);
#pragma unroll
        for (int j = 0; j < 8; j++) Vt[c + j][tid] = x[j];
      }
    }
    wait_async0();
    __syncthreads();  // Kbuf + Vt ready for all waves

    // S = Q K^T : 8 column tiles x (K-dim 64 -> 2 WMMA each), B frags from LDS
    v8f s[8];
#pragma unroll
    for (int j = 0; j < 8; j++) {
      const __bf16* kp = &Kbuf[16 * j + nl][hf * 8];
      v8f sj = {};
      v16bf bk = combine16(*(const v8bf*)kp, *(const v8bf*)(kp + 16));
      sj = WMMA_BF16(aq0, bk, sj);
      bk = combine16(*(const v8bf*)(kp + 32), *(const v8bf*)(kp + 48));
      sj = WMMA_BF16(aq1, bk, sj);
      s[j] = sj;
    }

    // Online softmax over 128 keys. Row m = v + 8*hf lives across the 16
    // lanes of this half-wave in vgpr v; one 4-level shuffle tree per row
    // per 128 keys (amortized 4x vs 32-key steps).
#pragma unroll
    for (int v = 0; v < 8; v++) {
      float sv[8];
#pragma unroll
      for (int j = 0; j < 8; j++) sv[j] = s[j][v] * scale;
      float rm = sv[0];
#pragma unroll
      for (int j = 1; j < 8; j++) rm = fmaxf(rm, sv[j]);
#pragma unroll
      for (int off = 8; off >= 1; off >>= 1)
        rm = fmaxf(rm, __shfl_xor(rm, off, 16));
      float mnew = fmaxf(mrun[v], rm);
      float alpha = __expf(mrun[v] - mnew);
      float rs = 0.f;
      int pr = v + 8 * hf;
#pragma unroll
      for (int j = 0; j < 8; j++) {
        float p = __expf(sv[j] - mnew);
        rs += p;
        Pl[wave][pr][16 * j + nl] = f2bf(p);
      }
#pragma unroll
      for (int off = 8; off >= 1; off >>= 1) rs += __shfl_xor(rs, off, 16);
      lrun[v] = lrun[v] * alpha + rs;
      mrun[v] = mnew;
#pragma unroll
      for (int j = 0; j < 4; j++) o[j][v] *= alpha;
    }
    __syncthreads();  // P visible (C-layout -> A-layout exchange via LDS)

    // O += P @ V : 4 K-chunks of 32; A frag from Pl, B frags from Vt
#pragma unroll
    for (int kc = 0; kc < 4; kc++) {
      const __bf16* pp = &Pl[wave][nl][kc * 32];
      v16bf ap = combine16(*(const v8bf*)(pp + hf * 8),
                           *(const v8bf*)(pp + hf * 8 + 16));
#pragma unroll
      for (int nt = 0; nt < 4; nt++) {
        const __bf16* vp = &Vt[nt * 16 + nl][kc * 32 + hf * 8];
        v16bf bv = combine16(*(const v8bf*)vp, *(const v8bf*)(vp + 16));
        o[nt] = WMMA_BF16(ap, bv, o[nt]);
      }
    }
  }

  // Epilogue: normalize and store flat [B, nh, T, hs]
#pragma unroll
  for (int v = 0; v < 8; v++) {
    float inv = 1.0f / lrun[v];
    int m = qbase + v + 8 * hf;
    size_t rowoff = (((size_t)b * NH + h) * T + m) * HS;
#pragma unroll
    for (int nt = 0; nt < 4; nt++)
      out[rowoff + nt * 16 + nl] = f2bf(o[nt][v] * inv);
  }
}

// ---------------------------------------------------------------------------
// Host launcher
// ---------------------------------------------------------------------------
extern "C" void kernel_launch(void* const* d_in, const int* in_sizes, int n_in,
                              void* d_out, int out_size, void* d_ws,
                              size_t ws_size, hipStream_t stream) {
  const float* x = (const float*)d_in[0];        // [4,2048,1024]
  const float* w_weight = (const float*)d_in[1]; // [3072,1024]
  const float* w_bias = (const float*)d_in[2];   // [3072]
  const float* proj_w = (const float*)d_in[3];   // [1024,1024]
  const float* proj_b = (const float*)d_in[4];   // [1024]
  float* out = (float*)d_out;                    // [4,2048,1024] f32

  const int B = 4, T = 2048, C = 1024, NH = 16;
  const long long M = (long long)B * T;  // 8192

  // Workspace carve-up (~88 MB total)
  char* ws = (char*)d_ws;
  __bf16* xb = (__bf16*)ws;   ws += (size_t)M * C * 2;        // 16 MB
  __bf16* wwb = (__bf16*)ws;  ws += (size_t)3 * C * C * 2;    //  6 MB
  __bf16* pwb = (__bf16*)ws;  ws += (size_t)C * C * 2;        //  2 MB
  __bf16* qkv = (__bf16*)ws;  ws += (size_t)M * 3 * C * 2;    // 48 MB
  __bf16* attn = (__bf16*)ws; ws += (size_t)M * C * 2;        // 16 MB

  f32_to_bf16_kernel<<<4096, 256, 0, stream>>>(x, xb, M * C);
  f32_to_bf16_kernel<<<2048, 256, 0, stream>>>(w_weight, wwb,
                                               (long long)3 * C * C);
  f32_to_bf16_kernel<<<1024, 256, 0, stream>>>(proj_w, pwb, (long long)C * C);

  // QKV projection: [8192,1024] x [3072,1024]^T -> bf16 [8192,3072]
  gemm_bias_wmma<true><<<dim3(3 * C / 128, M / 128), 256, 0, stream>>>(
      xb, wwb, w_bias, qkv, (int)M, 3 * C, C);

  // Fused attention
  attn_fused_wmma<<<dim3(T / 64, NH, B), 128, 0, stream>>>(qkv, attn);

  // Output projection: [8192,1024] x [1024,1024]^T -> f32 [8192,1024]
  gemm_bias_wmma<false><<<dim3(C / 128, M / 128), 256, 0, stream>>>(
      attn, pwb, proj_b, out, (int)M, C, C);
}
